// Model16_44495861186612
// MI455X (gfx1250) — matrix operand order repaired and checked by reading, not compiled, not run on hardware
//
#include <hip/hip_runtime.h>
#include <hip/hip_bf16.h>
#include <math.h>

typedef __attribute__((ext_vector_type(16))) __bf16 v16bf;
typedef __attribute__((ext_vector_type(8)))  __bf16 v8bf;
typedef __attribute__((ext_vector_type(8)))  float  v8f;

#define BN_SCALE 0.99999500003749975f

__device__ __forceinline__ unsigned short f2bf(float f) {
  unsigned int u = __builtin_bit_cast(unsigned int, f);
  u = (u + 0x7FFFu + ((u >> 16) & 1u)) >> 16;
  return (unsigned short)u;
}
__device__ __forceinline__ float sigmoidf_(float x) { return 1.0f / (1.0f + __expf(-x)); }

__device__ __forceinline__ v16bf mkv16(v8bf lo, v8bf hi) {
  return __builtin_shufflevector(lo, hi, 0, 1, 2, 3, 4, 5, 6, 7, 8, 9, 10, 11, 12, 13, 14, 15);
}
__device__ __forceinline__ v8f wmma_bf16(v16bf a, v16bf b, v8f c) {
  return __builtin_amdgcn_wmma_f32_16x16x32_bf16(false, a, false, b, (short)0, c, false, false);
}

// CDNA5 async global->LDS copy of 16 bytes (per lane), tracked by ASYNCcnt.
__device__ __forceinline__ void async_cp16(unsigned lds_off, const unsigned short* g) {
  asm volatile("global_load_async_to_lds_b128 %0, %1, off" ::"v"(lds_off), "v"(g) : "memory");
}
__device__ __forceinline__ void wait_async() {
  asm volatile("s_wait_asynccnt 0x0" ::: "memory");
}

// ---------------------------------------------------------------- f32 -> bf16
__global__ void k_f2bf(const float* __restrict__ in, unsigned short* __restrict__ out, int n) {
  int i = blockIdx.x * blockDim.x + threadIdx.x;
  if (i < n) out[i] = f2bf(in[i]);
}

// ------------------------------------------------------- conv0 (1->32ch, stride 2)
__global__ void k_conv0(const float* __restrict__ xx, const float* __restrict__ w,
                        const float* __restrict__ bias, float* __restrict__ out) {
  int i = blockIdx.x * blockDim.x + threadIdx.x;
  int wq = i & 127, h = (i >> 7) & 1023, o = (i >> 17) & 31, b = i >> 22;
  const float* xr = xx + ((size_t)b * 2048 + 2 * h) * 256;
  float acc = bias[o];
#pragma unroll
  for (int k = 0; k < 3; ++k) {
    int m = 2 * wq + k - 1;
    if (m >= 0 && m < 256) acc += xr[m] * w[o * 3 + k];
  }
  out[i] = acc;
}

// ----------------------------------- generic 32ch 1x3 conv (+resid) * BN_SCALE, silu
__global__ void k_conv32(const float* __restrict__ in, const float* __restrict__ w,
                         const float* __restrict__ bias, const float* __restrict__ resid,
                         float* __restrict__ out, int W) {
  __shared__ float wsh[3072];
  __shared__ float bsh[32];
  for (int j = threadIdx.x; j < 3072; j += blockDim.x) wsh[j] = w[j];
  if (threadIdx.x < 32) bsh[threadIdx.x] = bias[threadIdx.x];
  __syncthreads();
  int i = blockIdx.x * blockDim.x + threadIdx.x;
  int wq = i % W;
  int tmp = i / W;
  int h = tmp & 1023, o = (tmp >> 10) & 31, b = tmp >> 15;
  float acc = bsh[o];
  const float* base = in + (((size_t)b * 32) * 1024 + h) * (size_t)W + wq;
#pragma unroll 8
  for (int c = 0; c < 32; ++c) {
    const float* p = base + (size_t)c * 1024 * W;
    float x0 = (wq > 0) ? p[-1] : 0.f;
    float x1 = p[0];
    float x2 = (wq < W - 1) ? p[1] : 0.f;
    const float* ww = &wsh[(o * 32 + c) * 3];
    acc += x0 * ww[0] + x1 * ww[1] + x2 * ww[2];
  }
  if (resid) acc += resid[i];
  acc *= BN_SCALE;
  out[i] = acc * sigmoidf_(acc);
}

// ------------------------------------------------ maxpool window 3, stride 2, pad 1
__global__ void k_maxpool(const float* __restrict__ in, float* __restrict__ out, int W) {
  int W2 = W >> 1;
  int i = blockIdx.x * blockDim.x + threadIdx.x;
  int wq = i % W2;
  int rc = i / W2;
  const float* p = in + (size_t)rc * W + 2 * wq;
  float m = fmaxf(p[0], p[1]);
  if (wq > 0) m = fmaxf(m, p[-1]);
  out[i] = m;
}

// -------------------- (B,32,1024,32) -> bf16 rows (b*1024+t, c*32+w) for proj GEMM
__global__ void k_flatten_bf16(const float* __restrict__ in, unsigned short* __restrict__ out) {
  int i = blockIdx.x * blockDim.x + threadIdx.x;
  int j = i & 1023, t = (i >> 10) & 1023, b = i >> 20;
  int c = j >> 5, w = j & 31;
  out[i] = f2bf(in[(((size_t)(b * 32 + c)) * 1024 + t) * 32 + w]);
}

__global__ void k_add_bf16(const float* __restrict__ a, const float* __restrict__ b,
                           unsigned short* __restrict__ out, int n) {
  int i = blockIdx.x * blockDim.x + threadIdx.x;
  if (i < n) out[i] = f2bf(a[i] + b[i]);
}

// ------------- Y2 (8192,1024) -> (B,32,1024,32) image layout (inverse of flatten)
__global__ void k_rearr(const float* __restrict__ in, float* __restrict__ out) {
  int i = blockIdx.x * blockDim.x + threadIdx.x;
  int w = i & 31, t = (i >> 5) & 1023, c = (i >> 15) & 31, b = i >> 20;
  out[i] = in[((size_t)(b * 1024 + t)) * 1024 + c * 32 + w];
}

// ----------------------------------------- convT 1x2 (32->32), width doubles
__global__ void k_convT(const float* __restrict__ in, const float* __restrict__ w,
                        const float* __restrict__ bias, float* __restrict__ out, int W) {
  __shared__ float wsh[2048];
  for (int j = threadIdx.x; j < 2048; j += blockDim.x) wsh[j] = w[j];
  __syncthreads();
  int Wout = W * 2;
  int i = blockIdx.x * blockDim.x + threadIdx.x;
  int wout = i % Wout;
  int tmp = i / Wout;
  int h = tmp & 1023, o = (tmp >> 10) & 31, b = tmp >> 15;
  int k = wout & 1, wq = wout >> 1;
  float acc = bias[o];
#pragma unroll 8
  for (int c = 0; c < 32; ++c)
    acc += in[(((size_t)(b * 32 + c)) * 1024 + h) * (size_t)W + wq] * wsh[(c * 32 + o) * 2 + k];
  out[i] = acc;
}

// ---------------- convT3 (32->1) + sigmoid + repeat(interval=2 on H) + channel 0
__global__ void k_final(const float* __restrict__ in, const float* __restrict__ w,
                        const float* __restrict__ bias, float* __restrict__ out) {
  int i = blockIdx.x * blockDim.x + threadIdx.x;
  int wout = i & 255, h = (i >> 8) & 1023, b = i >> 18;
  int k = wout & 1, wq = wout >> 1;
  float acc = bias[0];
#pragma unroll 8
  for (int c = 0; c < 32; ++c)
    acc += in[(((size_t)(b * 32 + c)) * 1024 + h) * 128 + wq] * w[c * 2 + k];
  float v = sigmoidf_(acc);
  size_t o0 = ((size_t)b * 2048 + 2 * h) * 256 + wout;
  out[o0] = v;
  out[o0 + 256] = v;
}

// ============================================================================
// bf16 WMMA GEMM with async global->LDS double buffering.
// C = A @ Wt^T (+bias0+bias1) (+= if accumulate). A (M,K), Wt (N,K) bf16 row-major.
// Block = 256 threads (8 waves) computes a 128(M) x 64(N) tile; each wave a 32x32
// patch as 2x2 WMMA accumulators. Per K-chunk (BK=32): A tile 128x32 (8KB) and
// B tile 64x32 (4KB) staged in LDS via GLOBAL_LOAD_ASYNC_TO_LDS_B128 (ASYNCcnt),
// double buffered so copies overlap the 4 WMMAs/wave of the current chunk.
// ============================================================================
#define GBM 128
#define GBN 64

__global__ void __launch_bounds__(256) k_gemm_lds(
    const unsigned short* __restrict__ A, const unsigned short* __restrict__ Wt,
    const float* __restrict__ bias0, const float* __restrict__ bias1,
    float* __restrict__ C, int M, int N, int K, int accumulate) {
  __shared__ __align__(16) unsigned short As[2][GBM * 32];  // 2 x 8KB
  __shared__ __align__(16) unsigned short Bs[2][GBN * 32];  // 2 x 4KB
  const int nbn = N / GBN;
  const int bm = blockIdx.x / nbn;
  const int bn = blockIdx.x - bm * nbn;
  const int tid = threadIdx.x;
  const int lane = tid & 31;
  const int wv = tid >> 5;
  const int lrow = lane & 15;
  const int hi = lane >> 4;
  const int wm = wv >> 1;  // 0..3  (32-row band)
  const int wn = wv & 1;   // 0..1  (32-col band)

  // this thread's 16B async-copy slots: A rows (r, r+64), one B row
  const int crow = tid >> 2;
  const int cc8 = (tid & 3) * 8;
  const unsigned short* gA0 = A + (size_t)(bm * GBM + crow) * K + cc8;
  const unsigned short* gA1 = A + (size_t)(bm * GBM + crow + 64) * K + cc8;
  const unsigned short* gB0 = Wt + (size_t)(bn * GBN + crow) * K + cc8;
  const unsigned la0 = (unsigned)(size_t)&As[0][crow * 32 + cc8];
  const unsigned la1 = (unsigned)(size_t)&As[0][(crow + 64) * 32 + cc8];
  const unsigned lb0 = (unsigned)(size_t)&Bs[0][crow * 32 + cc8];
  const unsigned strA = (unsigned)(GBM * 32 * 2);  // bytes between As[0] and As[1]
  const unsigned strB = (unsigned)(GBN * 32 * 2);

  v8f acc00 = {0, 0, 0, 0, 0, 0, 0, 0};
  v8f acc01 = acc00, acc10 = acc00, acc11 = acc00;

  // prologue: chunk 0 -> buffer 0
  async_cp16(la0, gA0);
  async_cp16(la1, gA1);
  async_cp16(lb0, gB0);
  wait_async();
  __syncthreads();

  const int nk = K >> 5;
  const int ar0 = (wm * 32 + lrow) * 32;
  const int ar1 = (wm * 32 + 16 + lrow) * 32;
  const int br0 = (wn * 32 + lrow) * 32;
  const int br1 = (wn * 32 + 16 + lrow) * 32;

  for (int k = 0; k < nk; ++k) {
    const int buf = k & 1;
    if (k + 1 < nk) {  // issue next chunk into the other buffer
      const size_t go = (size_t)(k + 1) * 32;
      const unsigned lo = (unsigned)(buf ^ 1);
      async_cp16(la0 + lo * strA, gA0 + go);
      async_cp16(la1 + lo * strA, gA1 + go);
      async_cp16(lb0 + lo * strB, gB0 + go);
    }
    const unsigned short* as = As[buf];
    const unsigned short* bs = Bs[buf];
    v16bf va0 = mkv16(*(const v8bf*)&as[ar0 + hi * 8], *(const v8bf*)&as[ar0 + hi * 8 + 16]);
    v16bf va1 = mkv16(*(const v8bf*)&as[ar1 + hi * 8], *(const v8bf*)&as[ar1 + hi * 8 + 16]);
    v16bf vb0 = mkv16(*(const v8bf*)&bs[br0 + hi * 8], *(const v8bf*)&bs[br0 + hi * 8 + 16]);
    v16bf vb1 = mkv16(*(const v8bf*)&bs[br1 + hi * 8], *(const v8bf*)&bs[br1 + hi * 8 + 16]);
    acc00 = wmma_bf16(va0, vb0, acc00);
    acc01 = wmma_bf16(va0, vb1, acc01);
    acc10 = wmma_bf16(va1, vb0, acc10);
    acc11 = wmma_bf16(va1, vb1, acc11);
    if (k + 1 < nk) wait_async();
    __syncthreads();
  }

  // store 2x2 tiles
#pragma unroll
  for (int mi = 0; mi < 2; ++mi) {
#pragma unroll
    for (int ni = 0; ni < 2; ++ni) {
      v8f acc = (mi == 0) ? (ni == 0 ? acc00 : acc01) : (ni == 0 ? acc10 : acc11);
      const int ncol = bn * GBN + wn * 32 + ni * 16 + lrow;
      float badd = 0.f;
      if (bias0) badd += bias0[ncol];
      if (bias1) badd += bias1[ncol];
#pragma unroll
      for (int r = 0; r < 8; ++r) {
        const int mrow = bm * GBM + wm * 32 + mi * 16 + r + hi * 8;
        const size_t idx = (size_t)mrow * N + ncol;
        float v = acc[r] + badd;
        if (accumulate) v += C[idx];
        C[idx] = v;
      }
    }
  }
}

// -------------------------------------------------------------- recurrent scan
// One persistent workgroup (32 waves). State s (8x1024) lives in LDS (f32 + bf16
// WMMA-A copy, zero-padded to 16 rows). Per step: U = s@jA^T, V = s@A^T via WMMA,
// dt = sigmoid(U + XJ[t]); s = tanh((XB[t] + V)*dt + s). Wave w owns cols [32w,32w+32).
__global__ void __launch_bounds__(1024) k_scan(
    const float* __restrict__ XJ, const float* __restrict__ XB,
    const unsigned short* __restrict__ WjA, const unsigned short* __restrict__ WA,
    const float* __restrict__ s0, int s0_bstride,
    float* __restrict__ OUT, int reverse) {
  __shared__ float sf[8 * 1024];
  __shared__ __align__(16) unsigned short sb[16 * 1024];
  const int tid = threadIdx.x;
  for (int i = tid; i < 8 * 1024; i += 1024) {
    int b = i >> 10, h = i & 1023;
    float v = s0[(size_t)b * s0_bstride + h];
    sf[i] = v;
    sb[i] = f2bf(v);
    sb[8 * 1024 + i] = 0;  // zero padding rows 8..15 of the A operand
  }
  __syncthreads();
  const int lane = tid & 31;
  const int wv = tid >> 5;
  const int lrow = lane & 15;
  const int hi = lane >> 4;
  const int n0 = wv * 32 + lrow;
  const int n1 = n0 + 16;
  const unsigned short* wj0 = WjA + (size_t)n0 * 1024;
  const unsigned short* wj1 = WjA + (size_t)n1 * 1024;
  const unsigned short* wa0 = WA + (size_t)n0 * 1024;
  const unsigned short* wa1 = WA + (size_t)n1 * 1024;
  for (int st = 0; st < 1024; ++st) {
    const int t = reverse ? (1023 - st) : st;
    v8f u0 = {0, 0, 0, 0, 0, 0, 0, 0};
    v8f u1 = u0, v0 = u0, v1 = u0;
    for (int kk = 0; kk < 1024; kk += 32) {
      const int kb = kk + hi * 16;
      v16bf va = *(const v16bf*)&sb[lrow * 1024 + kb];
      const int kbb = kk + hi * 16;
      v16bf bj0 = *(const v16bf*)(wj0 + kbb);
      v16bf bj1 = *(const v16bf*)(wj1 + kbb);
      v16bf ba0 = *(const v16bf*)(wa0 + kbb);
      v16bf ba1 = *(const v16bf*)(wa1 + kbb);
      u0 = wmma_bf16(va, bj0, u0);
      u1 = wmma_bf16(va, bj1, u1);
      v0 = wmma_bf16(va, ba0, v0);
      v1 = wmma_bf16(va, ba1, v1);
    }
    __syncthreads();  // all waves done reading sb before updates
    if (hi == 0) {    // lanes 0..15 hold batches 0..7 in acc rows r=0..7
#pragma unroll
      for (int r = 0; r < 8; ++r) {
        const size_t rowoff = ((size_t)(r * 1024 + t)) * 1024;
        {
          float dt = sigmoidf_(u0[r] + XJ[rowoff + n0]);
          float vv = v0[r] + XB[rowoff + n0];
          float sn = tanhf(vv * dt + sf[r * 1024 + n0]);
          sf[r * 1024 + n0] = sn;
          sb[r * 1024 + n0] = f2bf(sn);
          OUT[rowoff + n0] = sn;
        }
        {
          float dt = sigmoidf_(u1[r] + XJ[rowoff + n1]);
          float vv = v1[r] + XB[rowoff + n1];
          float sn = tanhf(vv * dt + sf[r * 1024 + n1]);
          sf[r * 1024 + n1] = sn;
          sb[r * 1024 + n1] = f2bf(sn);
          OUT[rowoff + n1] = sn;
        }
      }
    }
    __syncthreads();  // updates visible before next step's reads
  }
}

// ============================================================================
extern "C" void kernel_launch(void* const* d_in, const int* in_sizes, int n_in,
                              void* d_out, int out_size, void* d_ws, size_t ws_size,
                              hipStream_t stream) {
  const float* xx = (const float*)d_in[0];
  const float* state_forward = (const float*)d_in[2];
  const float* conv0_w = (const float*)d_in[3];
  const float* conv0_b = (const float*)d_in[4];
  const float* cres_w1 = (const float*)d_in[5];
  const float* cres_b1 = (const float*)d_in[6];
  const float* cres_w2 = (const float*)d_in[7];
  const float* cres_b2 = (const float*)d_in[8];
  const float* proj_w = (const float*)d_in[9];
  const float* jA_w = (const float*)d_in[10];
  const float* jA_b = (const float*)d_in[11];
  const float* jB_w = (const float*)d_in[12];
  const float* jB_b = (const float*)d_in[13];
  const float* A_w = (const float*)d_in[14];
  const float* A_b = (const float*)d_in[15];
  const float* B_w = (const float*)d_in[16];
  const float* B_b = (const float*)d_in[17];
  const float* C_w = (const float*)d_in[18];
  const float* C_b = (const float*)d_in[19];
  const float* D_w = (const float*)d_in[20];
  const float* D_b = (const float*)d_in[21];
  const float* proj2_w = (const float*)d_in[22];
  const float* ores_w1 = (const float*)d_in[23];
  const float* ores_b1 = (const float*)d_in[24];
  const float* ores_w2 = (const float*)d_in[25];
  const float* ores_b2 = (const float*)d_in[26];
  const float* ct1_w = (const float*)d_in[27];
  const float* ct1_b = (const float*)d_in[28];
  const float* ct2_w = (const float*)d_in[29];
  const float* ct2_b = (const float*)d_in[30];
  const float* ct3_w = (const float*)d_in[31];
  const float* ct3_b = (const float*)d_in[32];
  float* outp = (float*)d_out;

  char* ws = (char*)d_ws;
  const size_t S_BUF = (size_t)8 * 32 * 1024 * 128 * 4;  // 134 MB ping-pong image buffers
  float* BUF0 = (float*)(ws);
  float* BUF1 = (float*)(ws + S_BUF);
  size_t off = 2 * S_BUF;
  unsigned short* FEATBF = (unsigned short*)(ws + off); off += (size_t)8192 * 1024 * 2;
  unsigned short* WBF    = (unsigned short*)(ws + off); off += (size_t)8 * 1024 * 1024 * 2;
  float* XF  = (float*)(ws + off);          off += (size_t)8192 * 1024 * 4;
  unsigned short* XBF = (unsigned short*)(ws + off); off += (size_t)8192 * 1024 * 2;
  float* XJ  = (float*)(ws + off);          off += (size_t)8192 * 1024 * 4;
  float* XBv = (float*)(ws + off);          off += (size_t)8192 * 1024 * 4;
  float* FWD = (float*)(ws + off);          off += (size_t)8192 * 1024 * 4;
  float* BWD = (float*)(ws + off);          off += (size_t)8192 * 1024 * 4;
  unsigned short* SUMBF = (unsigned short*)(ws + off); off += (size_t)8192 * 1024 * 2;
  float* Y1  = (float*)(ws + off);          off += (size_t)8192 * 1024 * 4;
  unsigned short* Y1BF = (unsigned short*)(ws + off); off += (size_t)8192 * 1024 * 2;
  float* Y2  = (float*)(ws + off);          off += (size_t)8192 * 1024 * 4;

  const int T = 256;
  const size_t MM = 1024 * 1024;
  const int GEMM_BLOCKS = (8192 / GBM) * (1024 / GBN);  // 64 * 16 = 1024

  // weights -> bf16 (all (N,K) row-major, K contiguous = WMMA-B friendly)
  k_f2bf<<<(int)(MM / T), T, 0, stream>>>(proj_w,  WBF + 0 * MM, (int)MM);
  k_f2bf<<<(int)(MM / T), T, 0, stream>>>(jA_w,    WBF + 1 * MM, (int)MM);
  k_f2bf<<<(int)(MM / T), T, 0, stream>>>(jB_w,    WBF + 2 * MM, (int)MM);
  k_f2bf<<<(int)(MM / T), T, 0, stream>>>(A_w,     WBF + 3 * MM, (int)MM);
  k_f2bf<<<(int)(MM / T), T, 0, stream>>>(B_w,     WBF + 4 * MM, (int)MM);
  k_f2bf<<<(int)(MM / T), T, 0, stream>>>(C_w,     WBF + 5 * MM, (int)MM);
  k_f2bf<<<(int)(MM / T), T, 0, stream>>>(D_w,     WBF + 6 * MM, (int)MM);
  k_f2bf<<<(int)(MM / T), T, 0, stream>>>(proj2_w, WBF + 7 * MM, (int)MM);

  // ------- conv front-end
  k_conv0<<<131072, T, 0, stream>>>(xx, conv0_w, conv0_b, BUF0);
  for (int u = 0; u < 2; ++u) {  // W=128, state in BUF0
    k_conv32<<<131072, T, 0, stream>>>(BUF0, cres_w1 + u * 3072, cres_b1 + u * 32, nullptr, BUF1, 128);
    k_conv32<<<131072, T, 0, stream>>>(BUF1, cres_w2 + u * 3072, cres_b2 + u * 32, BUF0, BUF0, 128);
  }
  k_maxpool<<<65536, T, 0, stream>>>(BUF0, BUF1, 128);  // -> W=64, state in BUF1
  for (int u = 2; u < 4; ++u) {
    k_conv32<<<65536, T, 0, stream>>>(BUF1, cres_w1 + u * 3072, cres_b1 + u * 32, nullptr, BUF0, 64);
    k_conv32<<<65536, T, 0, stream>>>(BUF0, cres_w2 + u * 3072, cres_b2 + u * 32, BUF1, BUF1, 64);
  }
  k_maxpool<<<32768, T, 0, stream>>>(BUF1, BUF0, 64);  // -> W=32, state in BUF0
  k_flatten_bf16<<<32768, T, 0, stream>>>(BUF0, FEATBF);

  // ------- projections (bias of the paired linear folded in where summed)
  k_gemm_lds<<<GEMM_BLOCKS, T, 0, stream>>>(FEATBF, WBF + 0 * MM, nullptr, nullptr, XF, 8192, 1024, 1024, 0);
  k_f2bf<<<32768, T, 0, stream>>>(XF, XBF, 8192 * 1024);
  k_gemm_lds<<<GEMM_BLOCKS, T, 0, stream>>>(XBF, WBF + 2 * MM, jB_b, jA_b, XJ,  8192, 1024, 1024, 0);
  k_gemm_lds<<<GEMM_BLOCKS, T, 0, stream>>>(XBF, WBF + 4 * MM, B_b,  A_b,  XBv, 8192, 1024, 1024, 0);

  // ------- bidirectional recurrent scan
  k_scan<<<1, 1024, 0, stream>>>(XJ, XBv, WBF + 1 * MM, WBF + 3 * MM, state_forward, 0, FWD, 0);
  k_scan<<<1, 1024, 0, stream>>>(XJ, XBv, WBF + 1 * MM, WBF + 3 * MM, FWD + 1023 * 1024, 1024 * 1024, BWD, 1);

  // ------- y = (fwd+bwd)@C^T + C_b + inp@D^T + D_b ; proj2
  k_add_bf16<<<32768, T, 0, stream>>>(FWD, BWD, SUMBF, 8192 * 1024);
  k_gemm_lds<<<GEMM_BLOCKS, T, 0, stream>>>(SUMBF, WBF + 5 * MM, C_b, nullptr, Y1, 8192, 1024, 1024, 0);
  k_gemm_lds<<<GEMM_BLOCKS, T, 0, stream>>>(XBF,   WBF + 6 * MM, D_b, nullptr, Y1, 8192, 1024, 1024, 1);
  k_f2bf<<<32768, T, 0, stream>>>(Y1, Y1BF, 8192 * 1024);
  k_gemm_lds<<<GEMM_BLOCKS, T, 0, stream>>>(Y1BF, WBF + 7 * MM, nullptr, nullptr, Y2, 8192, 1024, 1024, 0);
  k_rearr<<<32768, T, 0, stream>>>(Y2, BUF0);  // -> (8,32,1024,32), state in BUF0

  // ------- conv back-end
  k_conv32<<<32768, T, 0, stream>>>(BUF0, ores_w1 + 0 * 3072, ores_b1 + 0,  nullptr, BUF1, 32);
  k_conv32<<<32768, T, 0, stream>>>(BUF1, ores_w2 + 0 * 3072, ores_b2 + 0,  BUF0, BUF0, 32);
  k_convT<<<65536, T, 0, stream>>>(BUF0, ct1_w, ct1_b, BUF1, 32);   // -> W=64
  k_conv32<<<65536, T, 0, stream>>>(BUF1, ores_w1 + 1 * 3072, ores_b1 + 32, nullptr, BUF0, 64);
  k_conv32<<<65536, T, 0, stream>>>(BUF0, ores_w2 + 1 * 3072, ores_b2 + 32, BUF1, BUF1, 64);
  k_convT<<<131072, T, 0, stream>>>(BUF1, ct2_w, ct2_b, BUF0, 64);  // -> W=128
  k_conv32<<<131072, T, 0, stream>>>(BUF0, ores_w1 + 2 * 3072, ores_b1 + 64, nullptr, BUF1, 128);
  k_conv32<<<131072, T, 0, stream>>>(BUF1, ores_w2 + 2 * 3072, ores_b2 + 64, BUF0, BUF0, 128);
  k_final<<<8192, T, 0, stream>>>(BUF0, ct3_w, ct3_b, outp);

  (void)in_sizes; (void)n_in; (void)out_size; (void)ws_size;
}
